// EntityClassify_33294586479271
// MI455X (gfx1250) — matrix operand order repaired and checked by reading, not compiled, not run on hardware
//
#include <hip/hip_runtime.h>

typedef __attribute__((ext_vector_type(16))) _Float16 v16h;
typedef __attribute__((ext_vector_type(8)))  float    v8f;

#define H_DIM   64
#define OUT_DIM 16
#define NREL    90
#define NBASES  8
#define LDS_STRIDE 20   // 16 + 4 pad: avoids LDS bank conflicts on the C transpose

// ---------------- counting sort of edges by etype (padded to 16) ----------------
__global__ void k_fill(int* __restrict__ perm, int epmax, int* __restrict__ hist) {
  int i = blockIdx.x * blockDim.x + threadIdx.x;
  if (i < epmax) perm[i] = -1;
  if (i < NREL)  hist[i] = 0;
}

__global__ void k_hist(const int* __restrict__ etype, int E, int* __restrict__ hist) {
  int i = blockIdx.x * blockDim.x + threadIdx.x;
  if (i < E) atomicAdd(&hist[etype[i]], 1);
}

__global__ void k_prefix(const int* __restrict__ hist, int* __restrict__ rel_off,
                         int* __restrict__ cursor) {
  int acc = 0;
  for (int r = 0; r < NREL; ++r) {
    rel_off[r] = acc;
    cursor[r]  = acc;
    acc += (hist[r] + 15) & ~15;   // pad each relation bucket to a multiple of 16
  }
  rel_off[NREL] = acc;
}

__global__ void k_scatter(const int* __restrict__ etype, int E,
                          int* __restrict__ cursor, int* __restrict__ perm) {
  int i = blockIdx.x * blockDim.x + threadIdx.x;
  if (i < E) {
    int p = atomicAdd(&cursor[etype[i]], 1);
    perm[p] = i;
  }
}

// ---------------- W_r = sum_b coeff[r,b] * bases[b], emitted as f16 ----------------
template <int HOUT>
__global__ void k_wcombine(const float* __restrict__ coeff, const float* __restrict__ bases,
                           _Float16* __restrict__ Wh) {
  int idx = blockIdx.x * blockDim.x + threadIdx.x;
  const int total = NREL * H_DIM * HOUT;
  if (idx >= total) return;
  int o = idx % HOUT;
  int t = idx / HOUT;
  int i = t % H_DIM;
  int r = t / H_DIM;
  float acc = 0.f;
#pragma unroll
  for (int b = 0; b < NBASES; ++b)
    acc += coeff[r * NBASES + b] * bases[(b * H_DIM + i) * HOUT + o];
  Wh[idx] = (_Float16)acc;   // layout: Wh[(r*H_DIM + k)*HOUT + o]  (k-row contiguous in o)
}

// ---------------- elementwise helpers ----------------
template <int HOUT>
__global__ void k_bias_init(const float* __restrict__ bias, float* __restrict__ out, int n) {
  int i = blockIdx.x * blockDim.x + threadIdx.x;
  if (i < n) out[i] = bias[i % HOUT];
}

__global__ void k_convert(const float* __restrict__ h, _Float16* __restrict__ xh, int n) {
  int i = blockIdx.x * blockDim.x + threadIdx.x;
  if (i < n) xh[i] = (_Float16)h[i];
}

__global__ void k_relu_convert(const float* __restrict__ h, _Float16* __restrict__ xh, int n) {
  int i = blockIdx.x * blockDim.x + threadIdx.x;
  if (i < n) xh[i] = (_Float16)fmaxf(h[i], 0.f);
}

// ---------------- WMMA edge kernel: one wave = one 16-edge tile ----------------
// A (16x32 f16, ISA layout): lane<16 -> row=lane, K in {0..7,16..23};
//                            lane>=16 -> row=lane-16, K in {8..15,24..31}
// B (32x16 f16): lane -> col n = lane&15, K rows in the same order as A
// C (16x16 f32): slot v -> row m = v + 8*(lane>=16), col n = lane&15
// Epilogue: transpose C through LDS so each lane owns one edge + 8 contiguous features.
template <int HOUT>
__global__ void k_edges(const _Float16* __restrict__ x, const _Float16* __restrict__ Wh,
                        const int* __restrict__ perm, const int* __restrict__ rel_off,
                        const int* __restrict__ src, const int* __restrict__ dst,
                        const float* __restrict__ norm, float* __restrict__ out,
                        int ntiles) {
  __shared__ float lds_t[8 * 16 * LDS_STRIDE];   // 8 waves/block, private slices

  const int lane = threadIdx.x & 31;
  const int wave = threadIdx.x >> 5;
  const int tile = blockIdx.x * (blockDim.x >> 5) + wave;
  if (tile >= ntiles) return;                 // wave-uniform
  const int base = tile * 16;

  const int e   = lane & 15;                  // edge row owned by this lane
  const int eid = perm[base + e];
  if (!__any(eid >= 0)) return;               // wave-uniform: fully padded tile

  // relation of this tile: binary search over padded offsets (uniform)
  int lo = 0, hi = NREL - 1;
  while (lo < hi) {
    int mid = (lo + hi + 1) >> 1;
    if (rel_off[mid] <= base) lo = mid; else hi = mid - 1;
  }
  const int r = lo;

  const bool  valid = (eid >= 0);
  const int   srow  = valid ? src[eid]  : 0;
  const int   drow  = valid ? dst[eid]  : 0;
  const float nrm   = valid ? norm[eid] : 0.f;
  float* orow = out + (size_t)drow * HOUT;

  // ---- A fragments (f16 feature row gather, 4 x 16B per lane) ----
  const _Float16* xr = x + (size_t)srow * H_DIM;
  const int kb = (lane < 16) ? 0 : 8;
  v16h a0, a1;
#pragma unroll
  for (int i = 0; i < 8; ++i) {
    a0[i]     = valid ? xr[kb + i]      : (_Float16)0.f;
    a0[8 + i] = valid ? xr[kb + 16 + i] : (_Float16)0.f;
    a1[i]     = valid ? xr[kb + 32 + i] : (_Float16)0.f;
    a1[8 + i] = valid ? xr[kb + 48 + i] : (_Float16)0.f;
  }

  const _Float16* wr0 = Wh + ((size_t)r * H_DIM + kb) * HOUT + (lane & 15);        // K rows 0..31, col n = lane&15
  const _Float16* wr1 = Wh + ((size_t)r * H_DIM + 32 + kb) * HOUT + (lane & 15);   // K rows 32..63, col n = lane&15

  float* lt = lds_t + wave * (16 * LDS_STRIDE);
  const int mrow = (lane >= 16) ? 8 : 0;      // C row offset owned by this lane half
  const int fb   = mrow;                      // feature half read back after transpose

#pragma unroll
  for (int nb = 0; nb < HOUT / 16; ++nb) {
    v16h b0, b1;
#pragma unroll
    for (int j = 0; j < 16; ++j) {
      b0[j] = wr0[((j < 8) ? j : j + 8) * HOUT + nb * 16];
      b1[j] = wr1[((j < 8) ? j : j + 8) * HOUT + nb * 16];
    }
    v8f c = {};
    c = __builtin_amdgcn_wmma_f32_16x16x32_f16(false, a0, false, b0, (short)0, c, false, false);
    c = __builtin_amdgcn_wmma_f32_16x16x32_f16(false, a1, false, b1, (short)0, c, false, false);

    // transpose C through LDS: lds[m][n], m = edge row, n = feature col
#pragma unroll
    for (int v = 0; v < 8; ++v)
      lt[(v + mrow) * LDS_STRIDE + (lane & 15)] = c[v];
    asm volatile("s_wait_dscnt 0x0" ::: "memory");   // cross-lane store->load fence

    float rbuf[8];
#pragma unroll
    for (int i = 0; i < 8; ++i)
      rbuf[i] = lt[e * LDS_STRIDE + fb + i];
    asm volatile("" ::: "memory");                   // keep loads before next chunk's stores

    if (valid) {
#pragma unroll
      for (int i = 0; i < 8; ++i)
        atomicAdd(&orow[nb * 16 + fb + i], rbuf[i] * nrm);
    }
  }
}

// ---------------- driver ----------------
extern "C" void kernel_launch(void* const* d_in, const int* in_sizes, int n_in,
                              void* d_out, int out_size, void* d_ws, size_t ws_size,
                              hipStream_t stream) {
  const float* feats  = (const float*)d_in[0];
  const float* coeff0 = (const float*)d_in[1];
  const float* bases0 = (const float*)d_in[2];
  const float* bias0  = (const float*)d_in[3];
  const float* coeff1 = (const float*)d_in[4];
  const float* bases1 = (const float*)d_in[5];
  const float* bias1  = (const float*)d_in[6];
  const float* coeff2 = (const float*)d_in[7];
  const float* bases2 = (const float*)d_in[8];
  const float* bias2  = (const float*)d_in[9];
  const int*   src    = (const int*)d_in[10];
  const int*   dst    = (const int*)d_in[11];
  const int*   etype  = (const int*)d_in[12];
  const float* norm   = (const float*)d_in[13];
  float* outp = (float*)d_out;

  const int N     = in_sizes[0] / H_DIM;
  const int E     = in_sizes[10];
  const int EPMAX = E + 16 * NREL;     // worst-case padded length (multiple of 16)

  // workspace carve-out (f32 first, then f16, then ints)
  char* ws = (char*)d_ws;
  float*    h_acc = (float*)ws;      ws += (size_t)N * H_DIM * sizeof(float);
  _Float16* xhA   = (_Float16*)ws;   ws += (size_t)N * H_DIM * sizeof(_Float16);
  _Float16* xhB   = (_Float16*)ws;   ws += (size_t)N * H_DIM * sizeof(_Float16);
  _Float16* Wh    = (_Float16*)ws;   ws += (size_t)NREL * H_DIM * H_DIM * sizeof(_Float16);
  int* perm    = (int*)ws;           ws += (size_t)EPMAX * sizeof(int);
  int* hist    = (int*)ws;           ws += NREL * sizeof(int);
  int* cursor  = (int*)ws;           ws += NREL * sizeof(int);
  int* rel_off = (int*)ws;           ws += (NREL + 1) * sizeof(int);

  const int T   = 256;
  const int nHW = N * H_DIM;

  // sort edges by relation (fixed across all 3 layers)
  k_fill   <<<(EPMAX + T - 1) / T, T, 0, stream>>>(perm, EPMAX, hist);
  k_hist   <<<(E + T - 1) / T,     T, 0, stream>>>(etype, E, hist);
  k_prefix <<<1, 1, 0, stream>>>(hist, rel_off, cursor);
  k_scatter<<<(E + T - 1) / T,     T, 0, stream>>>(etype, E, cursor, perm);

  const int ntiles = EPMAX / 16;
  const int egrid  = (ntiles + 7) / 8;          // 8 waves per 256-thread block

  // f16 mirror of the input features
  k_convert<<<(nHW + T - 1) / T, T, 0, stream>>>(feats, xhA, nHW);

  // ---- layer 0: xhA -> h_acc -> xhB (H->H, relu) ----
  k_wcombine<H_DIM><<<(NREL * H_DIM * H_DIM + T - 1) / T, T, 0, stream>>>(coeff0, bases0, Wh);
  k_bias_init<H_DIM><<<(nHW + T - 1) / T, T, 0, stream>>>(bias0, h_acc, nHW);
  k_edges<H_DIM><<<egrid, T, 0, stream>>>(xhA, Wh, perm, rel_off, src, dst, norm, h_acc, ntiles);
  k_relu_convert<<<(nHW + T - 1) / T, T, 0, stream>>>(h_acc, xhB, nHW);

  // ---- layer 1: xhB -> h_acc -> xhA (H->H, relu) ----
  k_wcombine<H_DIM><<<(NREL * H_DIM * H_DIM + T - 1) / T, T, 0, stream>>>(coeff1, bases1, Wh);
  k_bias_init<H_DIM><<<(nHW + T - 1) / T, T, 0, stream>>>(bias1, h_acc, nHW);
  k_edges<H_DIM><<<egrid, T, 0, stream>>>(xhB, Wh, perm, rel_off, src, dst, norm, h_acc, ntiles);
  k_relu_convert<<<(nHW + T - 1) / T, T, 0, stream>>>(h_acc, xhA, nHW);

  // ---- layer 2: xhA -> d_out (H->OUT, no relu) ----
  k_wcombine<OUT_DIM><<<(NREL * H_DIM * OUT_DIM + T - 1) / T, T, 0, stream>>>(coeff2, bases2, Wh);
  k_bias_init<OUT_DIM><<<(N * OUT_DIM + T - 1) / T, T, 0, stream>>>(bias2, outp, N * OUT_DIM);
  k_edges<OUT_DIM><<<egrid, T, 0, stream>>>(xhA, Wh, perm, rel_off, src, dst, norm, outp, ntiles);
}
